// MultiHeadAttention_13864154431545
// MI455X (gfx1250) — matrix operand order repaired and checked by reading, not compiled, not run on hardware
//
#include <hip/hip_runtime.h>

// ---------------------------------------------------------------------------
// MHA forward for MI455X (gfx1250, wave32): WMMA bf16 16x16x32 f32-accum +
// double-buffered GLOBAL_LOAD_ASYNC_TO_LDS_B128 staging (ASYNCcnt pipeline).
// S=2048, B=2, D=1024, H=16, DK=64
// ---------------------------------------------------------------------------
#define S_LEN  2048
#define BATCH  2
#define DMODEL 1024
#define NHEAD  16
#define DKH    64
#define JW     (S_LEN / 32)   // packed mask words per row

typedef __attribute__((ext_vector_type(16))) __bf16 v16bf;
typedef __attribute__((ext_vector_type(8)))  float  v8f;

union ABFrag { unsigned int u[8]; v16bf v; };

__device__ __forceinline__ unsigned short f2bf(float x) {
  unsigned int u = __float_as_uint(x);
  return (unsigned short)((u + 0x7FFFu + ((u >> 16) & 1u)) >> 16);   // RNE
}

// A-matrix (16x32, 16-bit) K-pair base for VGPR v, lane-half h (contiguous k = 16*half + 2*v, matches B)
__device__ __forceinline__ int a_k0(int v, int half) {
  return (half << 4) + (v << 1);
}

// v_permlane16_b32 xor-butterfly max across each 16-lane row (pure VALU)
__device__ __forceinline__ float permlane16_f(float x, unsigned s1, unsigned s2) {
  unsigned xi = (unsigned)__float_as_int(x);
  unsigned r = __builtin_amdgcn_permlane16(xi, xi, s1, s2, false, false);
  return __int_as_float((int)r);
}
__device__ __forceinline__ float rowmax16(float x) {
  x = fmaxf(x, permlane16_f(x, 0x67452301u, 0xEFCDAB89u));  // ^1
  x = fmaxf(x, permlane16_f(x, 0x54761032u, 0xDCFE98BAu));  // ^2
  x = fmaxf(x, permlane16_f(x, 0x32107654u, 0xBA98FEDCu));  // ^4
  x = fmaxf(x, permlane16_f(x, 0xFEDCBA98u, 0x76543210u));  // ^8
  return x;
}

// ---- CDNA5 async global->LDS copy (ASYNCcnt), per-lane 16B ----------------
__device__ __forceinline__ void async_b128(unsigned lds_off, const void* gptr) {
  unsigned long long ga = (unsigned long long)(uintptr_t)gptr;
  asm volatile("global_load_async_to_lds_b128 %0, %1, off"
               :: "v"(lds_off), "v"(ga) : "memory");
}
__device__ __forceinline__ unsigned lds_off(const void* p) {
  return (unsigned)(uintptr_t)p;   // generic LDS addr = {aperture_hi, offset_lo}
}
__device__ __forceinline__ void wait_async_0() { asm volatile("s_wait_asynccnt 0x0" ::: "memory"); }
__device__ __forceinline__ void wait_async_4() { asm volatile("s_wait_asynccnt 0x4" ::: "memory"); }
__device__ __forceinline__ void wait_async_8() { asm volatile("s_wait_asynccnt 0x8" ::: "memory"); }

// ---------------------------------------------------------------------------
// float -> packed bf16 (4 elements / thread)
// ---------------------------------------------------------------------------
__global__ void cvt4_bf16_kernel(const float* __restrict__ in,
                                 unsigned short* __restrict__ out, int n4) {
  int i = blockIdx.x * blockDim.x + threadIdx.x;
  if (i < n4) {
    float4 f = ((const float4*)in)[i];
    uint2 p;
    p.x = (unsigned int)f2bf(f.x) | ((unsigned int)f2bf(f.y) << 16);
    p.y = (unsigned int)f2bf(f.z) | ((unsigned int)f2bf(f.w) << 16);
    ((uint2*)out)[i] = p;
  }
}

// ---------------------------------------------------------------------------
// fp32 W[K][N] -> bf16 wT[N][K] (LDS tile transpose, coalesced both sides)
// ---------------------------------------------------------------------------
__global__ __launch_bounds__(128) void cvtT_bf16_kernel(
    const float* __restrict__ W, unsigned short* __restrict__ wT, int K, int N) {
  __shared__ __align__(16) unsigned short T[64 * 64];
  const int tid = threadIdx.x;
  const int k0 = blockIdx.y * 64, n0 = blockIdx.x * 64;
#pragma unroll
  for (int i = 0; i < 8; ++i) {
    int idx = tid + i * 128;          // 1024 float4 chunks
    int r = idx >> 4, c4 = idx & 15;
    float4 f = *(const float4*)(W + (size_t)(k0 + r) * N + n0 + c4 * 4);
    T[(c4 * 4 + 0) * 64 + r] = f2bf(f.x);
    T[(c4 * 4 + 1) * 64 + r] = f2bf(f.y);
    T[(c4 * 4 + 2) * 64 + r] = f2bf(f.z);
    T[(c4 * 4 + 3) * 64 + r] = f2bf(f.w);
  }
  __syncthreads();
#pragma unroll
  for (int i = 0; i < 4; ++i) {
    int idx = tid + i * 128;          // 512 uint4 chunks
    int n = idx >> 3, ch = idx & 7;
    *(uint4*)(wT + (size_t)(n0 + n) * K + k0 + ch * 8) =
        *(const uint4*)(&T[n * 64 + ch * 8]);
  }
}

// ---------------------------------------------------------------------------
// bf16 vp[(j*B+b)*D + h*64 + f] -> vT[((b*H+h)*64 + f)*S + j]  (key-contiguous)
// ---------------------------------------------------------------------------
__global__ __launch_bounds__(128) void vT_kernel(
    const unsigned short* __restrict__ vp, unsigned short* __restrict__ vT) {
  __shared__ __align__(16) unsigned short T[64 * 64];
  const int tid = threadIdx.x;
  const int j0 = blockIdx.x * 64;
  const int b = blockIdx.y & (BATCH - 1), h = blockIdx.y >> 1;
#pragma unroll
  for (int i = 0; i < 4; ++i) {
    int idx = tid + i * 128;          // 512 uint4 chunks in
    int r = idx >> 3, ch = idx & 7;   // key row, feat chunk
    uint4 d = *(const uint4*)(vp + ((size_t)(j0 + r) * BATCH + b) * DMODEL + h * DKH + ch * 8);
    const unsigned short* e = (const unsigned short*)&d;
#pragma unroll
    for (int q = 0; q < 8; ++q) T[(ch * 8 + q) * 64 + r] = e[q];
  }
  __syncthreads();
#pragma unroll
  for (int i = 0; i < 4; ++i) {
    int idx = tid + i * 128;          // 512 uint4 chunks out
    int f = idx >> 3, ch = idx & 7;
    *(uint4*)(vT + (((size_t)b * NHEAD + h) * DKH + f) * S_LEN + j0 + ch * 8) =
        *(const uint4*)(&T[f * 64 + ch * 8]);
  }
}

// ---------------------------------------------------------------------------
// pack int32 mask -> 1 bit per key via wave32 ballot (512 KB total)
// ---------------------------------------------------------------------------
__global__ void mask_pack_kernel(const int* __restrict__ mask,
                                 unsigned int* __restrict__ mbits) {
  int gid = blockIdx.x * blockDim.x + threadIdx.x;
  unsigned long long b = __ballot(mask[gid] != 0);
  if ((threadIdx.x & 31) == 0) mbits[gid >> 5] = (unsigned int)b;
}

// ---------------------------------------------------------------------------
// out[M,N] = (A[M,K](bf16) @ wT[N,K](bf16)^T + bias) * scale ; out bf16/f32
// 128 threads, tile 64x64, k-step 64, async double-buffered staging
// ---------------------------------------------------------------------------
template <int STORE_F32>
__global__ __launch_bounds__(128) void wmma_gemm_bias_kernel(
    const unsigned short* __restrict__ A, const unsigned short* __restrict__ W,
    const float* __restrict__ bias, void* __restrict__ out,
    int M, int N, int K, float scale) {
  __shared__ __align__(16) unsigned short A_lds[2][64 * 64];   // [row][k]
  __shared__ __align__(16) unsigned short Bt_lds[2][64 * 64];  // [n][k]

  const int tid = threadIdx.x;
  const int w = tid >> 5, lane = tid & 31;
  const int half = lane >> 4, ln = lane & 15;
  const int n0 = blockIdx.x * 64;
  const int m0 = blockIdx.y * 64;

  auto issue_stage = [&](int k0s, int buf) {
#pragma unroll
    for (int c = 0; c < 4; ++c) {
      int ci = tid * 4 + c;
      int r = ci >> 3, ch = ci & 7;
      async_b128(lds_off(&A_lds[buf][r * 64 + ch * 8]),
                 A + (size_t)(m0 + r) * K + k0s + ch * 8);
      async_b128(lds_off(&Bt_lds[buf][r * 64 + ch * 8]),
                 W + (size_t)(n0 + r) * K + k0s + ch * 8);
    }
  };

  v8f acc[4];
#pragma unroll
  for (int t = 0; t < 4; ++t)
#pragma unroll
    for (int e = 0; e < 8; ++e) acc[t][e] = 0.f;

  issue_stage(0, 0);
  int cur = 0;
  for (int k0 = 0; k0 < K; k0 += 64) {
    if (k0 + 64 < K) { issue_stage(k0 + 64, cur ^ 1); wait_async_8(); }
    else             { wait_async_0(); }
    __syncthreads();

#pragma unroll
    for (int kk = 0; kk < 2; ++kk) {
      ABFrag af;
#pragma unroll
      for (int v = 0; v < 8; ++v)
        af.u[v] = *(const unsigned int*)(
            &A_lds[cur][(w * 16 + ln) * 64 + kk * 32 + a_k0(v, half)]);
#pragma unroll
      for (int t = 0; t < 4; ++t) {
        ABFrag bfr;
#pragma unroll
        for (int v = 0; v < 8; ++v)
          bfr.u[v] = *(const unsigned int*)(
              &Bt_lds[cur][(t * 16 + ln) * 64 + kk * 32 + half * 16 + 2 * v]);
        acc[t] = __builtin_amdgcn_wmma_f32_16x16x32_bf16(
            false, af.v, false, bfr.v, (short)0, acc[t], false, false);
      }
    }
    __syncthreads();
    cur ^= 1;
  }

#pragma unroll
  for (int t = 0; t < 4; ++t) {
    float bv = bias[n0 + t * 16 + ln];
#pragma unroll
    for (int r = 0; r < 8; ++r) {
      int m = m0 + w * 16 + r + half * 8;
      float val = (acc[t][r] + bv) * scale;
      size_t idx = (size_t)m * N + n0 + t * 16 + ln;
      if (STORE_F32) ((float*)out)[idx] = val;
      else           ((unsigned short*)out)[idx] = f2bf(val);
    }
  }
}

// ---------------------------------------------------------------------------
// Flash attention: grid (S/64, B*H), 128 threads (4 waves, 16 q-rows each)
// 9 WMMAs / 32-key step; K/V tiles async double-buffered; max via permlane16;
// row sums via extra P @ ones WMMA tile.
// ---------------------------------------------------------------------------
__global__ __launch_bounds__(128) void flash_attn_kernel(
    const unsigned short* __restrict__ qb, const unsigned short* __restrict__ kb,
    const unsigned short* __restrict__ vT, const unsigned int* __restrict__ mbits,
    unsigned short* __restrict__ xb) {
  __shared__ __align__(16) unsigned short K_lds[2][32 * 64];   // [key][feat]
  __shared__ __align__(16) unsigned short Vt_lds[2][64 * 32];  // [feat][key]
  __shared__ __align__(16) unsigned short P_lds[4][16 * 32];

  const int tid = threadIdx.x;
  const int w = tid >> 5, lane = tid & 31;
  const int half = lane >> 4, ln = lane & 15;
  const int b = blockIdx.y & (BATCH - 1);
  const int h = blockIdx.y >> 1;
  const int i0 = (blockIdx.x * 4 + w) * 16;

  auto issue_tile = [&](int j0s, int buf) {
#pragma unroll
    for (int c = 0; c < 2; ++c) {
      int ci = tid * 2 + c;
      {
        int kr = ci >> 3, ch = ci & 7;          // 256 chunks of K tile
        async_b128(lds_off(&K_lds[buf][kr * 64 + ch * 8]),
                   kb + ((size_t)(j0s + kr) * BATCH + b) * DMODEL + h * DKH + ch * 8);
      }
      {
        int f = ci >> 2, ch = ci & 3;           // 256 chunks of V^T tile
        async_b128(lds_off(&Vt_lds[buf][f * 32 + ch * 8]),
                   vT + (((size_t)b * NHEAD + h) * DKH + f) * S_LEN + j0s + ch * 8);
      }
    }
  };

  // Q A-fragments (1/sqrt(dk) folded into q projection)
  ABFrag aq[2];
  {
    const unsigned short* qrow =
        qb + ((size_t)(i0 + ln) * BATCH + b) * DMODEL + h * DKH;
#pragma unroll
    for (int kk = 0; kk < 2; ++kk)
#pragma unroll
      for (int v = 0; v < 8; ++v)
        aq[kk].u[v] = *(const unsigned int*)(qrow + kk * 32 + a_k0(v, half));
  }

  ABFrag ones;   // B-matrix of 1.0 (layout-independent) for row sums
#pragma unroll
  for (int v = 0; v < 8; ++v) ones.u[v] = 0x3F803F80u;

  float m_r[8];
  v8f acc[5];    // 4 output tiles + row-sum tile
#pragma unroll
  for (int r = 0; r < 8; ++r) m_r[r] = -1e30f;
#pragma unroll
  for (int t = 0; t < 5; ++t)
#pragma unroll
    for (int e = 0; e < 8; ++e) acc[t][e] = 0.f;

  issue_tile(0, 0);
  int cur = 0;
  for (int jt = 0; jt < S_LEN / 32; ++jt) {
    const int j0 = jt * 32;
    if (jt + 1 < S_LEN / 32) { issue_tile(j0 + 32, cur ^ 1); wait_async_4(); }
    else                     { wait_async_0(); }
    __syncthreads();

    // ---- scores: two 16x16 subtiles over the 32 keys ----
    v8f sc[2];
#pragma unroll
    for (int s2 = 0; s2 < 2; ++s2) {
#pragma unroll
      for (int e = 0; e < 8; ++e) sc[s2][e] = 0.f;
#pragma unroll
      for (int kk = 0; kk < 2; ++kk) {
        ABFrag bk;
#pragma unroll
        for (int v = 0; v < 8; ++v)
          bk.u[v] = *(const unsigned int*)(
              &K_lds[cur][(s2 * 16 + ln) * 64 + kk * 32 + half * 16 + 2 * v]);
        sc[s2] = __builtin_amdgcn_wmma_f32_16x16x32_bf16(
            false, aq[kk].v, false, bk.v, (short)0, sc[s2], false, false);
      }
    }

    // ---- online softmax (max via permlane16, sums via WMMA) ----
    float alpha[8];
#pragma unroll
    for (int r = 0; r < 8; ++r) {
      unsigned bits = mbits[(i0 + r + half * 8) * JW + jt];
      float s0 = ((bits >> ln) & 1u)        ? sc[0][r] : -3.0e38f;
      float s1 = ((bits >> (16 + ln)) & 1u) ? sc[1][r] : -3.0e38f;
      float rmax = rowmax16(fmaxf(s0, s1));
      float newm = fmaxf(m_r[r], rmax);
      alpha[r] = __expf(m_r[r] - newm);
      m_r[r] = newm;
      float p0 = __expf(s0 - newm);
      float p1 = __expf(s1 - newm);
      int prow = r + half * 8;
      P_lds[w][prow * 32 + ln]      = f2bf(p0);
      P_lds[w][prow * 32 + 16 + ln] = f2bf(p1);
    }
#pragma unroll
    for (int t = 0; t < 5; ++t)
#pragma unroll
      for (int r = 0; r < 8; ++r) acc[t][r] *= alpha[r];

    __syncthreads();  // P visible before PV reads

    // ---- PV: P(16x32) @ V(32x64), plus P @ ones for row sums ----
    ABFrag ap;
#pragma unroll
    for (int v = 0; v < 8; ++v)
      ap.u[v] = *(const unsigned int*)(&P_lds[w][ln * 32 + a_k0(v, half)]);
#pragma unroll
    for (int t = 0; t < 4; ++t) {
      ABFrag bv;
#pragma unroll
      for (int v = 0; v < 8; ++v)
        bv.u[v] = *(const unsigned int*)(
            &Vt_lds[cur][(t * 16 + ln) * 32 + half * 16 + 2 * v]);
      acc[t] = __builtin_amdgcn_wmma_f32_16x16x32_bf16(
          false, ap.v, false, bv.v, (short)0, acc[t], false, false);
    }
    acc[4] = __builtin_amdgcn_wmma_f32_16x16x32_bf16(
        false, ap.v, false, ones.v, (short)0, acc[4], false, false);

    __syncthreads();  // all LDS reads done before next-next restage
    cur ^= 1;
  }

#pragma unroll
  for (int r = 0; r < 8; ++r) {
    float inv = 1.0f / acc[4][r];    // row-sum tile is lane-replicated
    int irow = i0 + r + half * 8;
    size_t base = ((size_t)irow * BATCH + b) * DMODEL + h * DKH;
#pragma unroll
    for (int t = 0; t < 4; ++t)
      xb[base + t * 16 + ln] = f2bf(acc[t][r] * inv);
  }
}

// ---------------------------------------------------------------------------
extern "C" void kernel_launch(void* const* d_in, const int* in_sizes, int n_in,
                              void* d_out, int out_size, void* d_ws, size_t ws_size,
                              hipStream_t stream) {
  (void)in_sizes; (void)n_in; (void)out_size; (void)ws_size;
  const float* query = (const float*)d_in[0];
  const float* key_i = (const float*)d_in[1];
  const float* value = (const float*)d_in[2];
  const int*   mask  = (const int*)d_in[3];
  const float* Wq = (const float*)d_in[4];
  const float* bq = (const float*)d_in[5];
  const float* Wk = (const float*)d_in[6];
  const float* bk = (const float*)d_in[7];
  const float* Wv = (const float*)d_in[8];
  const float* bv = (const float*)d_in[9];
  const float* Wo = (const float*)d_in[10];
  const float* bo = (const float*)d_in[11];
  float* out = (float*)d_out;

  const size_t MTOK = (size_t)S_LEN * BATCH * DMODEL;  // 4096*1024
  const size_t WSZ  = (size_t)DMODEL * DMODEL;

  unsigned short* ws   = (unsigned short*)d_ws;
  unsigned short* q_in = ws;  ws += MTOK;
  unsigned short* k_in = ws;  ws += MTOK;
  unsigned short* v_in = ws;  ws += MTOK;
  unsigned short* wq_t = ws;  ws += WSZ;
  unsigned short* wk_t = ws;  ws += WSZ;
  unsigned short* wv_t = ws;  ws += WSZ;
  unsigned short* wo_t = ws;  ws += WSZ;
  unsigned short* qp   = ws;  ws += MTOK;
  unsigned short* kp   = ws;  ws += MTOK;
  unsigned short* vp   = ws;  ws += MTOK;
  unsigned short* vt   = ws;  ws += MTOK;
  unsigned short* xb   = ws;  ws += MTOK;
  unsigned int*   mbits = (unsigned int*)ws;           // + 512 KB

  auto cvt = [&](const float* src, unsigned short* dst, size_t n) {
    int n4 = (int)(n / 4);
    cvt4_bf16_kernel<<<(n4 + 255) / 256, 256, 0, stream>>>(src, dst, n4);
  };
  cvt(query, q_in, MTOK);
  cvt(key_i, k_in, MTOK);
  cvt(value, v_in, MTOK);

  dim3 gt(DMODEL / 64, DMODEL / 64);
  cvtT_bf16_kernel<<<gt, 128, 0, stream>>>(Wq, wq_t, DMODEL, DMODEL);
  cvtT_bf16_kernel<<<gt, 128, 0, stream>>>(Wk, wk_t, DMODEL, DMODEL);
  cvtT_bf16_kernel<<<gt, 128, 0, stream>>>(Wv, wv_t, DMODEL, DMODEL);
  cvtT_bf16_kernel<<<gt, 128, 0, stream>>>(Wo, wo_t, DMODEL, DMODEL);

  mask_pack_kernel<<<(S_LEN * S_LEN) / 256, 256, 0, stream>>>(mask, mbits);

  const int M = S_LEN * BATCH, N = DMODEL, K = DMODEL;
  dim3 gg(N / 64, M / 64);  // (16, 64)
  wmma_gemm_bias_kernel<0><<<gg, 128, 0, stream>>>(q_in, wq_t, bq, qp, M, N, K, 0.125f);
  wmma_gemm_bias_kernel<0><<<gg, 128, 0, stream>>>(k_in, wk_t, bk, kp, M, N, K, 1.0f);
  wmma_gemm_bias_kernel<0><<<gg, 128, 0, stream>>>(v_in, wv_t, bv, vp, M, N, K, 1.0f);

  vT_kernel<<<dim3(S_LEN / 64, BATCH * NHEAD), 128, 0, stream>>>(vp, vt);

  dim3 ga(S_LEN / 64, BATCH * NHEAD);  // (32, 32)
  flash_attn_kernel<<<ga, 128, 0, stream>>>(qp, kp, vt, mbits, xb);

  wmma_gemm_bias_kernel<1><<<gg, 128, 0, stream>>>(xb, wo_t, bo, out, M, N, K, 1.0f);
}
